// SquashedGaussianMoEActor_5188320494123
// MI455X (gfx1250) — compile-verified, matrix operand order checked by reading, not run on hardware
//
#include <hip/hip_runtime.h>

#define B_SZ   8192
#define OBS_SZ 128
#define H_SZ   512
#define D_SZ   512
#define E_SZ   16
#define T_SZ   50
#define A_SZ   32

typedef __attribute__((ext_vector_type(16))) __bf16 v16bf;
typedef __attribute__((ext_vector_type(8)))  __bf16 v8bf;
typedef __attribute__((ext_vector_type(8)))  float  v8f;

#define WMMA_BF16(A, Bf, C) \
    __builtin_amdgcn_wmma_f32_16x16x32_bf16(false, (A), false, (Bf), (short)0, (C), false, false)

// ---------------------------------------------------------------------------
// f32 -> bf16 conversion (grid-stride) — used for the activation input (obs)
// ---------------------------------------------------------------------------
__global__ __launch_bounds__(256)
void cvt_f32_bf16(const float* __restrict__ in, __bf16* __restrict__ out, size_t n) {
    size_t i      = (size_t)blockIdx.x * blockDim.x + threadIdx.x;
    size_t stride = (size_t)gridDim.x * blockDim.x;
    for (; i < n; i += stride) out[i] = (__bf16)in[i];
}

// ---------------------------------------------------------------------------
// Pack a row-major f32 weight W[K,N] into bf16 WMMA B-fragment order:
//   out[((ntile*KT + kt)*2 + half)*256 + lane*8 + jj]
//     = W[kt*32 + kof(v,kh) + lo, ntile*16 + (lane&15)]
// where j = half*8+jj, v=j>>1, lo=j&1, kh=lane>>4,
//       kof(v,kh) = (v&4)*4 + (v&3)*2 + kh*8   (ISA 7.12.2, 16-bit 16x32)
// A wave's fragment load becomes two fully-contiguous 512B b128 loads.
// blockIdx.y = batch (experts); W/out advance by K*N per batch.
// ---------------------------------------------------------------------------
__global__ __launch_bounds__(256)
void pack_w_bf16(const float* __restrict__ Wbase, __bf16* __restrict__ outBase,
                 int K, int N) {
    const size_t per = (size_t)K * N;
    const float* W  = Wbase + per * blockIdx.y;
    __bf16*     out = outBase + per * blockIdx.y;
    const int KT = K >> 5;
    size_t idx    = (size_t)blockIdx.x * blockDim.x + threadIdx.x;
    size_t stride = (size_t)gridDim.x * blockDim.x;
    for (; idx < per; idx += stride) {
        const int jj    = (int)(idx & 7);
        const int lane  = (int)((idx >> 3) & 31);
        const int half  = (int)((idx >> 8) & 1);
        const size_t rem = idx >> 9;
        const int kt    = (int)(rem % KT);
        const int ntile = (int)(rem / KT);
        const int j  = (half << 3) + jj;
        const int v  = j >> 1;
        const int lo = j & 1;
        const int kh = lane >> 4;
        const int n  = (ntile << 4) + (lane & 15);
        const int kk = ((v & 4) << 2) + ((v & 3) << 1) + (kh << 3) + lo;
        const int k  = (kt << 5) + kk;
        out[idx] = (__bf16)W[(size_t)k * N + n];
    }
}

// ---------------------------------------------------------------------------
// Fragment loaders from a running pointer (immediate-offset friendly)
// ---------------------------------------------------------------------------
__device__ __forceinline__ v16bf ldA(const __bf16* __restrict__ p) {
    const v8bf lo = *(const v8bf*)(p);
    const v8bf hi = *(const v8bf*)(p + 16);
    return __builtin_shufflevector(lo, hi, 0,1,2,3,4,5,6,7,8,9,10,11,12,13,14,15);
}
__device__ __forceinline__ v16bf ldB(const __bf16* __restrict__ p) {
    const v8bf lo = *(const v8bf*)(p);
    const v8bf hi = *(const v8bf*)(p + 256);
    return __builtin_shufflevector(lo, hi, 0,1,2,3,4,5,6,7,8,9,10,11,12,13,14,15);
}

// ---------------------------------------------------------------------------
// Batched bf16 GEMM via V_WMMA_F32_16X16X32_BF16, B pre-packed.
// Block: 256 threads = 8 waves arranged 4(M) x 2(N); each wave owns a
// 32x64 tile = 2 A fragments x 4 B fragments = 8 WMMAs per 32-wide K step.
// Fast path: software-pipelined, unguarded loop with 6 running pointers —
// prefetch loads use constant immediate offsets (+64B A, +1024B B), address
// update is one v_add_u64 per pointer per K-step.
// Slow path (N edge, only the tiny N=32 head GEMMs): guarded simple loop.
// ---------------------------------------------------------------------------
template<bool RELU, bool OUTF32>
__global__ __launch_bounds__(256)
void wmma_gemm_bf16(const __bf16* __restrict__ Abase,
                    const __bf16* __restrict__ Bpacked,
                    void*         __restrict__ Cbase,
                    const float*  __restrict__ biasBase,
                    int M, int N, int K,
                    long lda, long ldc,
                    long strideA, long strideB, long strideC, long strideBias)
{
    const int batch = blockIdx.z;
    const __bf16* Am  = Abase + (size_t)batch * strideA;
    const __bf16* Bm  = Bpacked + (size_t)batch * strideB;
    const float* bias = biasBase ? (biasBase + (size_t)batch * strideBias) : nullptr;

    const int wave  = threadIdx.x >> 5;
    const int lane  = threadIdx.x & 31;
    const int m16   = lane & 15;
    const int kh    = lane >> 4;
    const int waveM = wave & 3;
    const int waveN = wave >> 2;
    const int row0  = blockIdx.y * 128 + waveM * 32;
    const int nbase = blockIdx.x * 128 + waveN * 64;
    const int KT    = K >> 5;

    const v8f vz = {0.f, 0.f, 0.f, 0.f, 0.f, 0.f, 0.f, 0.f};
    v8f acc[2][4] = {{vz, vz, vz, vz}, {vz, vz, vz, vz}};

    const __bf16* arow0 = Am + (size_t)(row0 + m16) * lda + (kh << 3);
    const __bf16* arow1 = arow0 + (size_t)16 * lda;

    if (nbase + 64 <= N) {
        // ---------------- fast path: full 32x64 tile, double-buffered ----------------
        const size_t nt0 = (size_t)(nbase >> 4);
        const __bf16* ap0 = arow0;
        const __bf16* ap1 = arow1;
        const __bf16* bp0 = Bm + (((size_t)(nt0 + 0) * KT) << 9) + (lane << 3);
        const __bf16* bp1 = Bm + (((size_t)(nt0 + 1) * KT) << 9) + (lane << 3);
        const __bf16* bp2 = Bm + (((size_t)(nt0 + 2) * KT) << 9) + (lane << 3);
        const __bf16* bp3 = Bm + (((size_t)(nt0 + 3) * KT) << 9) + (lane << 3);

        v16bf a0 = ldA(ap0);
        v16bf a1 = ldA(ap1);
        v16bf b0 = ldB(bp0);
        v16bf b1 = ldB(bp1);
        v16bf b2 = ldB(bp2);
        v16bf b3 = ldB(bp3);
        for (int kt = 0; kt < KT - 1; ++kt) {
            // prefetch next K-step (constant immediate offsets off running pointers)
            const v16bf a0n = ldA(ap0 + 32);
            const v16bf a1n = ldA(ap1 + 32);
            const v16bf b0n = ldB(bp0 + 512);
            const v16bf b1n = ldB(bp1 + 512);
            const v16bf b2n = ldB(bp2 + 512);
            const v16bf b3n = ldB(bp3 + 512);
            acc[0][0] = WMMA_BF16(a0, b0, acc[0][0]);
            acc[1][0] = WMMA_BF16(a1, b0, acc[1][0]);
            acc[0][1] = WMMA_BF16(a0, b1, acc[0][1]);
            acc[1][1] = WMMA_BF16(a1, b1, acc[1][1]);
            acc[0][2] = WMMA_BF16(a0, b2, acc[0][2]);
            acc[1][2] = WMMA_BF16(a1, b2, acc[1][2]);
            acc[0][3] = WMMA_BF16(a0, b3, acc[0][3]);
            acc[1][3] = WMMA_BF16(a1, b3, acc[1][3]);
            ap0 += 32; ap1 += 32;
            bp0 += 512; bp1 += 512; bp2 += 512; bp3 += 512;
            a0 = a0n; a1 = a1n; b0 = b0n; b1 = b1n; b2 = b2n; b3 = b3n;
        }
        acc[0][0] = WMMA_BF16(a0, b0, acc[0][0]);
        acc[1][0] = WMMA_BF16(a1, b0, acc[1][0]);
        acc[0][1] = WMMA_BF16(a0, b1, acc[0][1]);
        acc[1][1] = WMMA_BF16(a1, b1, acc[1][1]);
        acc[0][2] = WMMA_BF16(a0, b2, acc[0][2]);
        acc[1][2] = WMMA_BF16(a1, b2, acc[1][2]);
        acc[0][3] = WMMA_BF16(a0, b3, acc[0][3]);
        acc[1][3] = WMMA_BF16(a1, b3, acc[1][3]);
    } else {
        // ---------------- slow path: N-edge (N=32 head GEMMs only) ----------------
        for (int kt = 0; kt < KT; ++kt) {
            const int k0 = kt << 5;
            const v16bf a0 = ldA(arow0 + k0);
            const v16bf a1 = ldA(arow1 + k0);
#pragma unroll
            for (int nt = 0; nt < 4; ++nt) {
                const int col0 = nbase + nt * 16;
                if (col0 < N) {
                    const __bf16* bp = Bm + (((size_t)(col0 >> 4) * KT + kt) << 9) + (lane << 3);
                    const v16bf bfv = ldB(bp);
                    acc[0][nt] = WMMA_BF16(a0, bfv, acc[0][nt]);
                    acc[1][nt] = WMMA_BF16(a1, bfv, acc[1][nt]);
                }
            }
        }
    }

    // epilogue: bias + activation + store (C/D layout: lane n=L&15, VGPR r -> M=(L>>4)*8+r)
#pragma unroll
    for (int mt = 0; mt < 2; ++mt) {
#pragma unroll
        for (int nt = 0; nt < 4; ++nt) {
            const int col0 = nbase + nt * 16;
            if (col0 < N) {
                const int n   = col0 + m16;
                const float bv = bias ? bias[n] : 0.0f;
#pragma unroll
                for (int r = 0; r < 8; ++r) {
                    const int m = row0 + (mt << 4) + (kh << 3) + r;
                    float v = acc[mt][nt][r] + bv;
                    if (RELU) v = fmaxf(v, 0.0f);
                    if (OUTF32) {
                        float* Cm = (float*)Cbase + (size_t)batch * strideC;
                        Cm[(size_t)m * ldc + n] = v;
                    } else {
                        __bf16* Cm = (__bf16*)Cbase + (size_t)batch * strideC;
                        Cm[(size_t)m * ldc + n] = (__bf16)v;
                    }
                }
            }
        }
    }
}

// ---------------------------------------------------------------------------
// KQ[e,t,i] = sum_j key_mats[e,i,j] * task_queries[t,j]   (tiny: 0.4 GFLOP)
// ---------------------------------------------------------------------------
__global__ __launch_bounds__(256)
void kq_kernel(const float* __restrict__ key, const float* __restrict__ tq,
               float* __restrict__ KQ) {
    const int idx = blockIdx.x * 256 + threadIdx.x;
    if (idx >= E_SZ * T_SZ * D_SZ) return;
    const int i = idx % D_SZ;
    const int t = (idx / D_SZ) % T_SZ;
    const int e = idx / (D_SZ * T_SZ);
    const float* krow = key + ((size_t)e * D_SZ + i) * D_SZ;
    const float* q    = tq + (size_t)t * D_SZ;
    float s = 0.f;
    for (int j = 0; j < D_SZ; ++j) s = fmaf(krow[j], q[j], s);
    KQ[idx] = s;
}

// ---------------------------------------------------------------------------
// Per batch row: logits[e] = eo2[b,e,:] . KQ[e,task[b],:], softmax over E,
// then wo[b,e,:] = w[e] * eo2[b,e,:]   (folded softmax weight for tower GEMM)
// 512 threads = 16 waves, one expert per wave; wave32 xor-shuffle reduction.
// ---------------------------------------------------------------------------
__global__ __launch_bounds__(512)
void moe_softmax_weight(const __bf16* __restrict__ eo2, const float* __restrict__ KQ,
                        const int* __restrict__ task, __bf16* __restrict__ wo) {
    __shared__ float slog[E_SZ];
    const int b    = blockIdx.x;
    const int wave = threadIdx.x >> 5;
    const int lane = threadIdx.x & 31;
    const int tsk  = task[b];

    const __bf16* row = eo2 + (size_t)b * E_SZ * D_SZ + (size_t)wave * D_SZ;
    const float*  kq  = KQ + ((size_t)wave * T_SZ + tsk) * D_SZ;

    float s = 0.f;
    for (int i = lane; i < D_SZ; i += 32) s = fmaf((float)row[i], kq[i], s);
#pragma unroll
    for (int off = 16; off > 0; off >>= 1) s += __shfl_xor(s, off, 32);
    if (lane == 0) slog[wave] = s;
    __syncthreads();

    float mx = -3.4e38f;
#pragma unroll
    for (int e = 0; e < E_SZ; ++e) mx = fmaxf(mx, slog[e]);
    float den = 0.f;
#pragma unroll
    for (int e = 0; e < E_SZ; ++e) den += __expf(slog[e] - mx);
    const float w = __expf(slog[wave] - mx) / den;

    __bf16* wrow = wo + (size_t)b * E_SZ * D_SZ + (size_t)wave * D_SZ;
    for (int i = lane; i < D_SZ; i += 32) wrow[i] = (__bf16)(w * (float)row[i]);
}

// ---------------------------------------------------------------------------
// Squashed-Gaussian head: pi = tanh(mu + exp(clip(log_std))*eps),
// logp = Normal logprob + tanh correction, summed over A.
// ---------------------------------------------------------------------------
__global__ __launch_bounds__(256)
void head_kernel(const float* __restrict__ mu, const float* __restrict__ lstd,
                 const float* __restrict__ eps, float* __restrict__ pi,
                 float* __restrict__ logp) {
    const int b = blockIdx.x * 256 + threadIdx.x;
    if (b >= B_SZ) return;
    const float HL2PI = 0.91893853320467274178f; // 0.5*log(2*pi)
    const float LN2   = 0.69314718055994530942f;
    float lp = 0.f;
#pragma unroll 4
    for (int a = 0; a < A_SZ; ++a) {
        const size_t idx = (size_t)b * A_SZ + a;
        const float m = mu[idx];
        const float l = fminf(fmaxf(lstd[idx], -20.f), 2.f);
        const float e = eps[idx];
        const float act = fmaf(__expf(l), e, m);
        lp += -0.5f * e * e - l - HL2PI;
        const float x  = -2.f * act;
        const float sp = (x > 20.f) ? x : log1pf(__expf(x));
        lp -= 2.f * (LN2 - act - sp);
        pi[idx] = tanhf(act);
    }
    logp[b] = lp;
}

// ---------------------------------------------------------------------------
extern "C" void kernel_launch(void* const* d_in, const int* in_sizes, int n_in,
                              void* d_out, int out_size, void* d_ws, size_t ws_size,
                              hipStream_t stream) {
    (void)in_sizes; (void)n_in; (void)out_size; (void)ws_size;

    // inputs (setup_inputs order)
    const float* obs   = (const float*)d_in[0];
    const int*   task  = (const int*)  d_in[1];
    const float* eps   = (const float*)d_in[2];
    const float* bW1   = (const float*)d_in[3];  const float* bb1 = (const float*)d_in[4];
    const float* bW2   = (const float*)d_in[5];  const float* bb2 = (const float*)d_in[6];
    const float* eW1   = (const float*)d_in[7];  const float* eb1 = (const float*)d_in[8];
    const float* eW2   = (const float*)d_in[9];  const float* eb2 = (const float*)d_in[10];
    const float* keyM  = (const float*)d_in[11];
    const float* valM  = (const float*)d_in[12];
    const float* tq    = (const float*)d_in[13];
    const float* mW1   = (const float*)d_in[14]; const float* mb1 = (const float*)d_in[15];
    const float* mW2   = (const float*)d_in[16]; const float* mb2 = (const float*)d_in[17];
    const float* lW1   = (const float*)d_in[18]; const float* lb1 = (const float*)d_in[19];
    const float* lW2   = (const float*)d_in[20]; const float* lb2 = (const float*)d_in[21];

    // workspace carving
    char*  ws  = (char*)d_ws;
    size_t off = 0;
    auto carve = [&](size_t bytes) -> void* {
        void* p = ws + off;
        off += (bytes + 255) & ~(size_t)255;
        return p;
    };
    __bf16* obs_bf = (__bf16*)carve((size_t)B_SZ * OBS_SZ * 2);
    __bf16* pbW1   = (__bf16*)carve((size_t)OBS_SZ * H_SZ * 2);
    __bf16* pbW2   = (__bf16*)carve((size_t)H_SZ * H_SZ * 2);
    __bf16* peW1   = (__bf16*)carve((size_t)E_SZ * H_SZ * D_SZ * 2);
    __bf16* peW2   = (__bf16*)carve((size_t)E_SZ * D_SZ * D_SZ * 2);
    __bf16* pval   = (__bf16*)carve((size_t)E_SZ * D_SZ * D_SZ * 2);
    __bf16* pmW1   = (__bf16*)carve((size_t)D_SZ * H_SZ * 2);
    __bf16* pmW2   = (__bf16*)carve((size_t)H_SZ * A_SZ * 2);
    __bf16* plW1   = (__bf16*)carve((size_t)D_SZ * H_SZ * 2);
    __bf16* plW2   = (__bf16*)carve((size_t)H_SZ * A_SZ * 2);
    __bf16* h1     = (__bf16*)carve((size_t)B_SZ * H_SZ * 2);
    __bf16* h      = (__bf16*)carve((size_t)B_SZ * H_SZ * 2);
    __bf16* eo     = (__bf16*)carve((size_t)B_SZ * E_SZ * D_SZ * 2);   // reused as wo
    __bf16* eo2    = (__bf16*)carve((size_t)B_SZ * E_SZ * D_SZ * 2);
    __bf16* tower  = (__bf16*)carve((size_t)B_SZ * D_SZ * 2);
    __bf16* t1m    = (__bf16*)carve((size_t)B_SZ * H_SZ * 2);
    __bf16* t1l    = (__bf16*)carve((size_t)B_SZ * H_SZ * 2);
    float*  KQ     = (float*)carve((size_t)E_SZ * T_SZ * D_SZ * 4);
    float*  mub    = (float*)carve((size_t)B_SZ * A_SZ * 4);
    float*  lsb    = (float*)carve((size_t)B_SZ * A_SZ * 4);
    __bf16* wo     = eo; // eo no longer needed once eo2 exists

    // activation conversion
    {
        size_t n = (size_t)B_SZ * OBS_SZ;
        int blocks = (int)((n + 255) / 256); if (blocks > 4096) blocks = 4096;
        cvt_f32_bf16<<<blocks, 256, 0, stream>>>(obs, obs_bf, n);
    }
    // weight packing into WMMA B-fragment order
    auto pack = [&](const float* src, __bf16* dst, int K, int N, int batch) {
        size_t per = (size_t)K * N;
        int blocks = (int)((per + 255) / 256); if (blocks > 2048) blocks = 2048;
        dim3 grid(blocks, batch, 1);
        pack_w_bf16<<<grid, 256, 0, stream>>>(src, dst, K, N);
    };
    pack(bW1, pbW1, OBS_SZ, H_SZ, 1);
    pack(bW2, pbW2, H_SZ, H_SZ, 1);
    pack(eW1, peW1, H_SZ, D_SZ, E_SZ);
    pack(eW2, peW2, D_SZ, D_SZ, E_SZ);
    pack(valM, pval, E_SZ * D_SZ, D_SZ, 1);   // value_flat: [E*D, D] single weight
    pack(mW1, pmW1, D_SZ, H_SZ, 1);
    pack(mW2, pmW2, H_SZ, A_SZ, 1);
    pack(lW1, plW1, D_SZ, H_SZ, 1);
    pack(lW2, plW2, H_SZ, A_SZ, 1);

    auto gemm = [&](int relu, int outf32,
                    const __bf16* Ap, const __bf16* Bp, void* Cp, const float* biasp,
                    int M, int N, int K, long lda, long ldc,
                    long sA, long sB, long sC, long sBias, int batch) {
        dim3 grid((N + 127) / 128, M / 128, batch);
        if (relu)
            wmma_gemm_bf16<true, false><<<grid, 256, 0, stream>>>(
                Ap, Bp, Cp, biasp, M, N, K, lda, ldc, sA, sB, sC, sBias);
        else if (outf32)
            wmma_gemm_bf16<false, true><<<grid, 256, 0, stream>>>(
                Ap, Bp, Cp, biasp, M, N, K, lda, ldc, sA, sB, sC, sBias);
        else
            wmma_gemm_bf16<false, false><<<grid, 256, 0, stream>>>(
                Ap, Bp, Cp, biasp, M, N, K, lda, ldc, sA, sB, sC, sBias);
    };

    // backbone
    gemm(1, 0, obs_bf, pbW1, h1, bb1, B_SZ, H_SZ, OBS_SZ, OBS_SZ, H_SZ, 0, 0, 0, 0, 1);
    gemm(1, 0, h1, pbW2, h, bb2, B_SZ, H_SZ, H_SZ, H_SZ, H_SZ, 0, 0, 0, 0, 1);

    // KQ[e,t,:] = key_mats[e] @ task_queries[t]
    kq_kernel<<<(E_SZ * T_SZ * D_SZ + 255) / 256, 256, 0, stream>>>(keyM, tq, KQ);

    // experts (batched over E via blockIdx.z)
    gemm(1, 0, h, peW1, eo, eb1, B_SZ, D_SZ, H_SZ,
         H_SZ, (long)E_SZ * D_SZ,
         0, (long)H_SZ * D_SZ, D_SZ, D_SZ, E_SZ);
    gemm(0, 0, eo, peW2, eo2, eb2, B_SZ, D_SZ, D_SZ,
         (long)E_SZ * D_SZ, (long)E_SZ * D_SZ,
         D_SZ, (long)D_SZ * D_SZ, D_SZ, D_SZ, E_SZ);

    // attention logits -> softmax -> folded weights (overwrites eo as wo)
    moe_softmax_weight<<<B_SZ, 512, 0, stream>>>(eo2, KQ, task, wo);

    // tower[b,:] = wo[b,(e,i)] @ value_flat[(e,i),j]   K = E*D = 8192
    gemm(0, 0, wo, pval, tower, nullptr, B_SZ, D_SZ, E_SZ * D_SZ,
         (long)E_SZ * D_SZ, D_SZ, 0, 0, 0, 0, 1);

    // mu / log_std towers
    gemm(1, 0, tower, pmW1, t1m, mb1, B_SZ, H_SZ, D_SZ, D_SZ, H_SZ, 0, 0, 0, 0, 1);
    gemm(1, 0, tower, plW1, t1l, lb1, B_SZ, H_SZ, D_SZ, D_SZ, H_SZ, 0, 0, 0, 0, 1);
    gemm(0, 1, t1m, pmW2, mub, mb2, B_SZ, A_SZ, H_SZ, H_SZ, A_SZ, 0, 0, 0, 0, 1);
    gemm(0, 1, t1l, plW2, lsb, lb2, B_SZ, A_SZ, H_SZ, H_SZ, A_SZ, 0, 0, 0, 0, 1);

    // squashed-Gaussian head -> d_out = [pi (B*A) | logp (B)]
    float* out_pi   = (float*)d_out;
    float* out_logp = out_pi + (size_t)B_SZ * A_SZ;
    head_kernel<<<(B_SZ + 255) / 256, 256, 0, stream>>>(mub, lsb, eps, out_pi, out_logp);
}
